// FeatureWiseCrossAttentionLSTM_26757646254494
// MI455X (gfx1250) — matrix operand order repaired and checked by reading, not compiled, not run on hardware
//
#include <hip/hip_runtime.h>

// ---------------------------------------------------------------------------
// FeatureWiseCrossAttentionLSTM for MI455X (gfx1250, wave32, WMMA, TDM)
//
// F=40 features, L=3 LSTM layers, H=128, T=512, B=32, 4H=512 gates.
//   * LSTM recurrence: 1 workgroup per feature, 512 threads = 16 waves.
//     gates(32x512) = [h(32x128) | x_prev(32x128)] @ Wcomb^T(512x256) per step
//     via v_wmma_f32_16x16x32_bf16. Wcomb bf16 resident in LDS (~264KB; CDNA5
//     320KB WGP LDS). Hidden-half B-fragments hoisted into registers.
//   * x_t tile (prev layer h at time t) DMA'd by the Tensor Data Mover into a
//     double-buffered LDS tile (TDM applies the bank-conflict row padding),
//     overlapped with compute; published via s_wait_tensorcnt + barrier.
//   * h_seq output streamed with global_store_async_from_lds_b128 (ASYNCcnt).
//   * Tiny attention stages + FC: VALU kernels (<<1% FLOPs).
// ---------------------------------------------------------------------------

typedef __attribute__((ext_vector_type(16))) __bf16 v16bf;
typedef __attribute__((ext_vector_type(8)))  float  v8f;
typedef __attribute__((ext_vector_type(4)))  unsigned int v4u;
typedef __attribute__((ext_vector_type(8)))  int v8i;
typedef __attribute__((ext_vector_type(4)))  int v4i;

#define F_  40
#define OF  8
#define OW  24
#define H   128
#define NH  4
#define G   5
#define B   32
#define T   512
#define NG  512   // 4*H gates

// LDS strides in bf16 elements (padded to dodge 64-bank conflicts)
#define WSTR 264  // 256 + 8
#define HSTR 136  // 128 + 8

#if __has_builtin(__builtin_amdgcn_tensor_load_to_lds)
#define HAVE_TDM 1
#else
#define HAVE_TDM 0
#endif

union FragU { uint4 u4[2]; v16bf v; };

__device__ __forceinline__ float sigmoidf_(float x) {
    return 1.0f / (1.0f + __expf(-x));
}

// Wave-level TDM issue: 2D tile, 32 rows x 128 bf16, row stride T*H elements,
// LDS padding 4 dwords after every 64 dwords (-> HSTR=136 bf16 rows).
__device__ __forceinline__ void tdm_load_tile(unsigned lds_off,
                                              unsigned long long gaddr) {
#if HAVE_TDM
    v4u g0;
    g0[0] = 1u;                                     // count=1 (valid user D#)
    g0[1] = lds_off;                                // lds_addr (bytes)
    g0[2] = (unsigned)gaddr;                        // global_addr[31:0]
    g0[3] = (unsigned)((gaddr >> 32) & 0x01FFFFFFull) | (2u << 30); // [56:32] | type=2

    v8i g1;
    g1[0] = (1 << 16)      // data_size = 2 bytes
          | (1 << 20)      // pad_enable
          | (5 << 22)      // pad_interval: 64 dwords (= one 128-bf16 row)
          | (3 << 25);     // pad_amount: 4 dwords (= 8 bf16 -> stride 136)
    g1[1] = (H & 0xFFFF) << 16;          // tensor_dim0[15:0] = 128
    g1[2] = (B & 0xFFFF) << 16;          // tensor_dim0 hi=0 | tensor_dim1 = 32
    g1[3] = (H & 0xFFFF) << 16;          // tensor_dim1 hi=0 | tile_dim0 = 128
    g1[4] = (B & 0xFFFF);                // tile_dim1 = 32 | tile_dim2 = 0
    g1[5] = (int)(T * H);                // tensor_dim0_stride lo (elements)
    g1[6] = 0;                           // stride hi | dim1_stride lo
    g1[7] = 0;

    v4i gz = {0, 0, 0, 0};
#if defined(__clang_major__) && (__clang_major__ >= 23)
    v8i gz8 = {0, 0, 0, 0, 0, 0, 0, 0};
    __builtin_amdgcn_tensor_load_to_lds(g0, g1, gz, gz, gz8, 0);
#else
    __builtin_amdgcn_tensor_load_to_lds(g0, g1, gz, gz, 0);
#endif
#else
    (void)lds_off; (void)gaddr;
#endif
}

__device__ __forceinline__ void async_store_b128(unsigned long long gaddr,
                                                 unsigned lds_off) {
    asm volatile("global_store_async_from_lds_b128 %0, %1, off"
                 :: "v"(gaddr), "v"(lds_off) : "memory");
}

__device__ __forceinline__ void wait_asynccnt0() {
    asm volatile("s_wait_asynccnt 0x0" ::: "memory");
}

// ---------------------------------------------------------------------------
// LSTM layer kernel: grid = F_ blocks, 512 threads (16 waves).
// ---------------------------------------------------------------------------
__global__ __launch_bounds__(512)
void lstm_layer_kernel(const float* __restrict__ x,        // (B,F,T) layer0 input
                       const float* __restrict__ w_ih0,    // (F,512)
                       const float* __restrict__ w_ih12,   // (2,F,512,128)
                       const float* __restrict__ w_hh,     // (3,F,512,128)
                       const float* __restrict__ b_ih,     // (3,F,512)
                       const float* __restrict__ b_hh,     // (3,F,512)
                       const __bf16* __restrict__ hseq_in, // (F,B,T,H) bf16
                       __bf16* __restrict__ hseq_out,      // (F,B,T,H) bf16
                       float* __restrict__ features,       // (B,F,H) f32
                       int layer)
{
    extern __shared__ char smem[];
    __bf16* Wc  = (__bf16*)smem;            // [512][WSTR] combined [Whh | Wih]
    __bf16* hb  = Wc  + NG * WSTR;          // [32][HSTR]  h_{t-1}
    __bf16* xtb = hb  + B  * HSTR;          // 2 x [32][HSTR] prev-layer h tile
    float*  xs  = (float*)(xtb + 2 * B * HSTR); // [32] layer0 scalar input

    const int f   = blockIdx.x;
    const int tid = threadIdx.x;
    const int KW  = (layer == 0) ? 128 : 256;

    // ---- Stage combined weights into LDS as bf16 ----
    const float* whh_f = w_hh + ((size_t)(layer * F_ + f)) * NG * H;
    const float* wih_f = (layer > 0)
        ? (w_ih12 + ((size_t)((layer - 1) * F_ + f)) * NG * H) : nullptr;
    for (int idx = tid; idx < NG * 256; idx += 512) {
        int n = idx >> 8;
        int k = idx & 255;
        if (k < KW) {
            float v = (k < 128) ? whh_f[n * H + k] : wih_f[n * H + (k - 128)];
            Wc[n * WSTR + k] = (__bf16)v;
        }
    }
    for (int idx = tid; idx < B * HSTR; idx += 512)
        hb[idx] = (__bf16)0.0f;
    if (tid < B) xs[tid] = 0.0f;
    __syncthreads();     // weights + h0 staged

    // ---- Per-wave tile assignment ----
    const int wv   = tid >> 5;      // wave id 0..15
    const int lane = tid & 31;
    const int m    = wv & 1;        // batch tile
    const int j    = wv >> 1;       // hidden tile
    const int sel  = lane >> 4;
    const int ncol = lane & 15;
    const int arow = m * 16 + ncol; // A-fragment source row (batch index)

    // bias (+ layer0 rank-1 input weight) per lane per gate
    float bias4[4], wv4[4];
#pragma unroll
    for (int g = 0; g < 4; ++g) {
        int n = g * H + j * 16 + ncol;
        bias4[g] = b_ih[(size_t)(layer * F_ + f) * NG + n]
                 + b_hh[(size_t)(layer * F_ + f) * NG + n];
        wv4[g]   = (layer == 0) ? w_ih0[(size_t)f * NG + n] : 0.0f;
    }

    // Hoist hidden-recurrence B-fragments (K=0..127) into registers:
    // 16 fragments x 8 VGPRs = 128 VGPRs, loop-invariant.
    v16bf wreg[4][4];
#pragma unroll
    for (int c = 0; c < 4; ++c) {
#pragma unroll
        for (int g = 0; g < 4; ++g) {
            int n = g * H + j * 16 + ncol;
            const __bf16* bp = Wc + n * WSTR + c * 32 + sel * 8;
            FragU fw;
            fw.u4[0] = *(const uint4*)bp;
            fw.u4[1] = *(const uint4*)(bp + 16);
            wreg[c][g] = fw.v;
        }
    }

    v8f creg;
#pragma unroll
    for (int r = 0; r < 8; ++r) creg[r] = 0.0f;

    const unsigned xtb_off[2] = {
        (unsigned)(size_t)(void*)xtb,
        (unsigned)(size_t)(void*)(xtb + B * HSTR)
    };
    const unsigned long long hin_base = (layer > 0)
        ? (unsigned long long)(size_t)(hseq_in + ((size_t)(f * B)) * T * H) : 0ull;

    // ---- Stage t = 0 inputs ----
    if (layer == 0) {
        if (tid < B) xs[tid] = x[((size_t)tid * F_ + f) * T + 0];
    } else {
#if HAVE_TDM
        if (wv == 0) tdm_load_tile(xtb_off[0], hin_base);   // row t=0
#else
        int b  = tid >> 4;
        int h0 = (tid & 15) * 8;
        const __bf16* src = hseq_in + (((size_t)(f * B + b)) * T + 0) * H + h0;
        *(uint4*)(xtb + b * HSTR + h0) = *(const uint4*)src;
#endif
    }
    __syncthreads();

    // ---- Time loop ----
    for (int t = 0; t < T; ++t) {
#if HAVE_TDM
        if (layer > 0 && wv == 0) {
            if (t + 1 < T) {
                // DMA next timestep's tile into the other buffer, then make
                // sure the tile for THIS step has landed (in-order TENSORcnt).
                tdm_load_tile(xtb_off[(t + 1) & 1],
                              hin_base + (unsigned long long)(t + 1) * H * 2);
                __builtin_amdgcn_s_wait_tensorcnt(1);
            } else {
                __builtin_amdgcn_s_wait_tensorcnt(0);
            }
        }
        __syncthreads();   // xtb[t&1] published to all waves
#endif
        const __bf16* xcur = xtb + (t & 1) * (B * HSTR);

        v8f acc[4];
#pragma unroll
        for (int g = 0; g < 4; ++g) {
            float bv = bias4[g];
            if (layer == 0) {
#pragma unroll
                for (int r = 0; r < 8; ++r) {
                    int b = m * 16 + r + 8 * sel;
                    acc[g][r] = bv + xs[b] * wv4[g];
                }
            } else {
#pragma unroll
                for (int r = 0; r < 8; ++r) acc[g][r] = bv;
            }
        }

        FragU fa, fb;
        // K = 0..127 : hidden recurrence, B-fragments from registers
#pragma unroll
        for (int c = 0; c < 4; ++c) {
            const __bf16* ap = hb + arow * HSTR + c * 32 + sel * 8;
            fa.u4[0] = *(const uint4*)ap;
            fa.u4[1] = *(const uint4*)(ap + 16);
#pragma unroll
            for (int g = 0; g < 4; ++g) {
                acc[g] = __builtin_amdgcn_wmma_f32_16x16x32_bf16(
                    false, fa.v, false, wreg[c][g], (short)0, acc[g], false, false);
            }
        }
        // K = 128..255 : fused input projection (A from TDM tile)
        if (layer > 0) {
#pragma unroll
            for (int c = 0; c < 4; ++c) {
                int k0 = c * 32;
                const __bf16* ap = xcur + arow * HSTR + k0 + sel * 8;
                fa.u4[0] = *(const uint4*)ap;
                fa.u4[1] = *(const uint4*)(ap + 16);
#pragma unroll
                for (int g = 0; g < 4; ++g) {
                    int n = g * H + j * 16 + ncol;
                    const __bf16* bp = Wc + n * WSTR + 128 + k0 + sel * 8;
                    fb.u4[0] = *(const uint4*)bp;
                    fb.u4[1] = *(const uint4*)(bp + 16);
                    acc[g] = __builtin_amdgcn_wmma_f32_16x16x32_bf16(
                        false, fa.v, false, fb.v, (short)0, acc[g], false, false);
                }
            }
        }

        // ---- elementwise LSTM cell update, fully in registers ----
        v8f hnew;
#pragma unroll
        for (int r = 0; r < 8; ++r) {
            float ig = sigmoidf_(acc[0][r]);
            float fg = sigmoidf_(acc[1][r]);
            float gg = tanhf(acc[2][r]);
            float og = sigmoidf_(acc[3][r]);
            float c2 = fg * creg[r] + ig * gg;
            creg[r]  = c2;
            hnew[r]  = og * tanhf(c2);
        }

        wait_asynccnt0();  // prior h-tile async stores fully drained (per wave)
        __syncthreads();   // all reads of hb(t-1) done; safe to overwrite

        // write new h tile to LDS (+ features at the end of the last layer)
#pragma unroll
        for (int r = 0; r < 8; ++r) {
            int bb = m * 16 + r + 8 * sel;
            int hc = j * 16 + ncol;
            hb[bb * HSTR + hc] = (__bf16)hnew[r];
            if (layer == 2 && t == T - 1)
                features[((size_t)bb * F_ + f) * H + hc] = hnew[r];
        }

        // stage inputs for t+1 (layer0 scalars / non-TDM fallback tile)
        if (t + 1 < T) {
            if (layer == 0) {
                if (tid < B) xs[tid] = x[((size_t)tid * F_ + f) * T + (t + 1)];
            }
#if !HAVE_TDM
            else {
                int bb = tid >> 4;
                int h0 = (tid & 15) * 8;
                const __bf16* src =
                    hseq_in + (((size_t)(f * B + bb)) * T + (t + 1)) * H + h0;
                *(uint4*)(xtb + ((t + 1) & 1) * (B * HSTR) + bb * HSTR + h0) =
                    *(const uint4*)src;
                if (t + 2 < T) __builtin_prefetch(src + H, 0, 1);
            }
#endif
        }
        __syncthreads();   // hb(t) + staged inputs visible everywhere

        // stream h(t) to global for the next layer: 16 wave-level async
        // LDS->global 128-bit stores instead of 4096 scalar b16 stores.
        if (layer < 2) {
            int bb2 = tid >> 4;
            int h02 = (tid & 15) * 8;
            unsigned lds_off =
                (unsigned)(size_t)(void*)(hb + bb2 * HSTR + h02);
            unsigned long long ga = (unsigned long long)(size_t)
                (hseq_out + (((size_t)(f * B + bb2)) * T + t) * H + h02);
            async_store_b128(ga, lds_off);
        }
    }
}

// ---------------------------------------------------------------------------
// Per-group MHA (seq=OF=8, NH=4, hd=32) + mean pool. grid = G*B, 128 threads.
// ---------------------------------------------------------------------------
__global__ __launch_bounds__(128)
void group_attn_kernel(const float* __restrict__ features, // (B,F,H)
                       const float* __restrict__ wqkv,     // (G,384,128)
                       const float* __restrict__ bqkv,     // (G,384)
                       const float* __restrict__ wo,       // (G,128,128)
                       const float* __restrict__ bo,       // (G,128)
                       float* __restrict__ pooled)         // (B,G,H)
{
    __shared__ float S[OF][H];
    __shared__ float QKV[OF][3 * H];
    __shared__ float ATT[NH][OF][OF];
    __shared__ float O[OF][H];
    __shared__ float P[OF][H];

    const int g = blockIdx.x / B;
    const int b = blockIdx.x % B;
    const int tid = threadIdx.x;

    for (int idx = tid; idx < OF * H; idx += 128) {
        int of = idx >> 7, h = idx & 127;
        S[of][h] = features[((size_t)b * F_ + g * OF + of) * H + h];
    }
    __syncthreads();

    for (int idx = tid; idx < OF * 3 * H; idx += 128) {
        int of = idx / (3 * H), n = idx % (3 * H);
        const float* wr = wqkv + ((size_t)g * 3 * H + n) * H;
        float s = bqkv[(size_t)g * 3 * H + n];
        for (int k = 0; k < H; ++k) s += S[of][k] * wr[k];
        QKV[of][n] = s;
    }
    __syncthreads();

    for (int idx = tid; idx < NH * OF * OF; idx += 128) {
        int nh = idx >> 6, qo = (idx >> 3) & 7, ko = idx & 7;
        float s = 0.0f;
        for (int d = 0; d < 32; ++d)
            s += QKV[qo][nh * 32 + d] * QKV[ko][H + nh * 32 + d];
        ATT[nh][qo][ko] = s * 0.17677669529663687f;   // 1/sqrt(32)
    }
    __syncthreads();

    if (tid < NH * OF) {
        int nh = tid >> 3, qo = tid & 7;
        float mx = -1e30f;
        for (int ko = 0; ko < OF; ++ko) mx = fmaxf(mx, ATT[nh][qo][ko]);
        float sum = 0.0f;
        for (int ko = 0; ko < OF; ++ko) {
            float e = __expf(ATT[nh][qo][ko] - mx);
            ATT[nh][qo][ko] = e;
            sum += e;
        }
        float inv = 1.0f / sum;
        for (int ko = 0; ko < OF; ++ko) ATT[nh][qo][ko] *= inv;
    }
    __syncthreads();

    for (int idx = tid; idx < OF * H; idx += 128) {
        int of = idx >> 7, n = idx & 127, nh = n >> 5, d = n & 31;
        float s = 0.0f;
        for (int ko = 0; ko < OF; ++ko)
            s += ATT[nh][of][ko] * QKV[ko][2 * H + nh * 32 + d];
        O[of][n] = s;
    }
    __syncthreads();

    for (int idx = tid; idx < OF * H; idx += 128) {
        int of = idx >> 7, h = idx & 127;
        const float* wr = wo + ((size_t)g * H + h) * H;
        float s = bo[(size_t)g * H + h];
        for (int k = 0; k < H; ++k) s += O[of][k] * wr[k];
        P[of][h] = s;
    }
    __syncthreads();

    if (tid < H) {
        float s = 0.0f;
        for (int of = 0; of < OF; ++of) s += P[of][tid];
        pooled[((size_t)b * G + g) * H + tid] = s * (1.0f / OF);
    }
}

// ---------------------------------------------------------------------------
// Final MHA (seq=G=5, only query index 0 matters) + FC. grid = B, 128 threads.
// ---------------------------------------------------------------------------
__global__ __launch_bounds__(128)
void final_attn_fc_kernel(const float* __restrict__ pooled, // (B,G,H)
                          const float* __restrict__ wqkv,   // (384,128)
                          const float* __restrict__ bqkv,   // (384)
                          const float* __restrict__ wo,     // (128,128)
                          const float* __restrict__ bo,     // (128)
                          const float* __restrict__ fc_w,   // (192,128)
                          const float* __restrict__ fc_b,   // (192)
                          float* __restrict__ out)          // (B,OF,OW)
{
    __shared__ float S[G][H];
    __shared__ float QKV[G][3 * H];
    __shared__ float ATT[NH][G];
    __shared__ float O[H];
    __shared__ float E[H];

    const int b = blockIdx.x;
    const int tid = threadIdx.x;

    for (int idx = tid; idx < G * H; idx += 128) {
        int gi = idx >> 7, h = idx & 127;
        S[gi][h] = pooled[((size_t)b * G + gi) * H + h];
    }
    __syncthreads();

    for (int idx = tid; idx < G * 3 * H; idx += 128) {
        int gi = idx / (3 * H), n = idx % (3 * H);
        const float* wr = wqkv + (size_t)n * H;
        float s = bqkv[n];
        for (int k = 0; k < H; ++k) s += S[gi][k] * wr[k];
        QKV[gi][n] = s;
    }
    __syncthreads();

    if (tid < NH * G) {
        int nh = tid / G, ko = tid % G;
        float s = 0.0f;
        for (int d = 0; d < 32; ++d)
            s += QKV[0][nh * 32 + d] * QKV[ko][H + nh * 32 + d];
        ATT[nh][ko] = s * 0.17677669529663687f;
    }
    __syncthreads();

    if (tid < NH) {
        float mx = -1e30f;
        for (int ko = 0; ko < G; ++ko) mx = fmaxf(mx, ATT[tid][ko]);
        float sum = 0.0f;
        for (int ko = 0; ko < G; ++ko) {
            float e = __expf(ATT[tid][ko] - mx);
            ATT[tid][ko] = e;
            sum += e;
        }
        float inv = 1.0f / sum;
        for (int ko = 0; ko < G; ++ko) ATT[tid][ko] *= inv;
    }
    __syncthreads();

    if (tid < H) {
        int nh = tid >> 5, d = tid & 31;
        float s = 0.0f;
        for (int ko = 0; ko < G; ++ko)
            s += ATT[nh][ko] * QKV[ko][2 * H + nh * 32 + d];
        O[tid] = s;
    }
    __syncthreads();

    if (tid < H) {
        const float* wr = wo + (size_t)tid * H;
        float s = bo[tid];
        for (int k = 0; k < H; ++k) s += O[k] * wr[k];
        E[tid] = s;
    }
    __syncthreads();

    for (int j = tid; j < OF * OW; j += 128) {
        const float* wr = fc_w + (size_t)j * H;
        float s = fc_b[j];
        for (int k = 0; k < H; ++k) s += E[k] * wr[k];
        out[(size_t)b * (OF * OW) + j] = s;
    }
}

// ---------------------------------------------------------------------------
extern "C" void kernel_launch(void* const* d_in, const int* in_sizes, int n_in,
                              void* d_out, int out_size, void* d_ws, size_t ws_size,
                              hipStream_t stream) {
    const float* x      = (const float*)d_in[0];
    const float* w_ih0  = (const float*)d_in[1];
    const float* w_ih12 = (const float*)d_in[2];
    const float* w_hh   = (const float*)d_in[3];
    const float* b_ih   = (const float*)d_in[4];
    const float* b_hh   = (const float*)d_in[5];
    const float* g_wqkv = (const float*)d_in[6];
    const float* g_bqkv = (const float*)d_in[7];
    const float* g_wo   = (const float*)d_in[8];
    const float* g_bo   = (const float*)d_in[9];
    const float* f_wqkv = (const float*)d_in[10];
    const float* f_bqkv = (const float*)d_in[11];
    const float* f_wo   = (const float*)d_in[12];
    const float* f_bo   = (const float*)d_in[13];
    const float* fc_w   = (const float*)d_in[14];
    const float* fc_b   = (const float*)d_in[15];
    float* out = (float*)d_out;

    // workspace carve-up
    char* ws = (char*)d_ws;
    const size_t hseq_elems = (size_t)F_ * B * T * H;   // bf16 elements
    __bf16* hseqA    = (__bf16*)ws;
    __bf16* hseqB    = (__bf16*)(ws + hseq_elems * 2);
    float*  features = (float*)(ws + hseq_elems * 4);
    float*  pooled   = (float*)(ws + hseq_elems * 4 + (size_t)B * F_ * H * 4);

    const size_t smem =
        (size_t)NG * WSTR * 2 +       // Wcomb bf16
        (size_t)B * HSTR * 2 +        // hb bf16
        (size_t)B * HSTR * 2 * 2 +    // xtb double buffer bf16
        (size_t)B * 4;                // xs f32

    lstm_layer_kernel<<<F_, 512, smem, stream>>>(
        x, w_ih0, w_ih12, w_hh, b_ih, b_hh, nullptr, hseqA, features, 0);
    lstm_layer_kernel<<<F_, 512, smem, stream>>>(
        x, w_ih0, w_ih12, w_hh, b_ih, b_hh, hseqA, hseqB, features, 1);
    lstm_layer_kernel<<<F_, 512, smem, stream>>>(
        x, w_ih0, w_ih12, w_hh, b_ih, b_hh, hseqB, hseqA, features, 2);

    group_attn_kernel<<<G * B, 128, 0, stream>>>(
        features, g_wqkv, g_bqkv, g_wo, g_bo, pooled);

    final_attn_fc_kernel<<<B, 128, 0, stream>>>(
        pooled, f_wqkv, f_bqkv, f_wo, f_bo, fc_w, fc_b, out);
}
